// ClassAttentionBlock_67697274519917
// MI455X (gfx1250) — matrix operand order repaired and checked by reading, not compiled, not run on hardware
//
#include <hip/hip_runtime.h>
#include <hip/hip_bf16.h>
#include <math.h>

#define DIM     384
#define NTOK    4097
#define BATCH   16
#define HEADS   8
#define HD      48
#define HIDDEN  1536
#define LN_EPS  1e-5f
#define ATTN_SCALE 0.14433756729740643f   // 48^-0.5
#define TILE    64
#define NTILES  65                        // ceil(4097/64)
#define NPAD    (NTILES*TILE)             // 4160
#define LDA     392                       // padded LDS row stride (bf16 elems); row = 784 B

typedef __attribute__((ext_vector_type(16))) __bf16 v16bf;
typedef __attribute__((ext_vector_type(8)))  __bf16 v8bf;
typedef __attribute__((ext_vector_type(8)))  float  v8f;
typedef __attribute__((ext_vector_type(4)))  unsigned int v4u;

#define WMMA_BF16(A, B, C) \
  __builtin_amdgcn_wmma_f32_16x16x32_bf16(false, (A), false, (B), (short)0, (C), false, false)

// ---------------- reductions ----------------
__device__ __forceinline__ float waveRedSum(float v) {
#pragma unroll
  for (int m = 16; m >= 1; m >>= 1) v += __shfl_xor(v, m, 32);
  return v;
}
__device__ __forceinline__ float halfRedSum(float v) {   // reduce within 16-lane halves
#pragma unroll
  for (int m = 8; m >= 1; m >>= 1) v += __shfl_xor(v, m, 32);
  return v;
}
__device__ __forceinline__ float blockRedSum(float v, float* red) {
  v = waveRedSum(v);
  const int wave = threadIdx.x >> 5;
  const int lane = threadIdx.x & 31;
  if (lane == 0) red[wave] = v;
  __syncthreads();
  if (threadIdx.x == 0) {
    float s = 0.f;
#pragma unroll
    for (int w = 0; w < 8; ++w) s += red[w];
    red[0] = s;
  }
  __syncthreads();
  float r = red[0];
  __syncthreads();
  return r;
}

__device__ __forceinline__ v16bf combineA(v4u lo, v4u hi) {
  v8bf l = __builtin_bit_cast(v8bf, lo);
  v8bf h = __builtin_bit_cast(v8bf, hi);
  return __builtin_shufflevector(l, h, 0,1,2,3,4,5,6,7,8,9,10,11,12,13,14,15);
}

// Pinned LDS loader: 4 A-fragments (one per m-tile) for one kc step.
// volatile => cannot be LICM-hoisted out of the o-tile loop (prevents the
// 384-VGPR / scratch-spill blowup); includes its own s_wait_dscnt since the
// waitcnt pass cannot see inside the asm. No memory clobber: aLDS is written
// strictly before __syncthreads and read-only afterwards, so global-load
// scheduling around this block stays legal.
__device__ __forceinline__ void loadA4(unsigned int addr,
                                       v16bf& a0, v16bf& a1, v16bf& a2, v16bf& a3) {
  v4u l0, h0, l1, h1, l2, h2, l3, h3;
  asm volatile(
      "ds_load_b128 %0, %8\n\t"
      "ds_load_b128 %1, %8 offset:32\n\t"
      "ds_load_b128 %2, %8 offset:12544\n\t"
      "ds_load_b128 %3, %8 offset:12576\n\t"
      "ds_load_b128 %4, %8 offset:25088\n\t"
      "ds_load_b128 %5, %8 offset:25120\n\t"
      "ds_load_b128 %6, %8 offset:37632\n\t"
      "ds_load_b128 %7, %8 offset:37664\n\t"
      "s_wait_dscnt 0x0"
      : "=v"(l0), "=v"(h0), "=v"(l1), "=v"(h1),
        "=v"(l2), "=v"(h2), "=v"(l3), "=v"(h3)
      : "v"(addr));
  a0 = combineA(l0, h0);
  a1 = combineA(l1, h1);
  a2 = combineA(l2, h2);
  a3 = combineA(l3, h3);
}

// ---------------- kernel: convert K/V weights to bf16 ----------------
__global__ __launch_bounds__(256) void convw_kernel(const float* __restrict__ qkv_w,
                                                    __bf16* __restrict__ Wkv) {
  int i = blockIdx.x * 256 + threadIdx.x;
  if (i < 768 * DIM) Wkv[i] = (__bf16)qkv_w[DIM * DIM + i];  // rows 384..1151
}

// ---------------- kernel: class-token query (f32, lane-parallel dots) ----------------
__global__ __launch_bounds__(256) void qcls_kernel(const float* __restrict__ x,
                                                   const float* __restrict__ ln1w,
                                                   const float* __restrict__ ln1b,
                                                   const float* __restrict__ qkv_w,
                                                   float* __restrict__ qcls) {
  const int b = blockIdx.x;
  __shared__ float a_s[DIM];
  __shared__ float red[8];
  const int wave = threadIdx.x >> 5;
  const int lane = threadIdx.x & 31;
  const float* xr = x + (size_t)b * NTOK * DIM;
  float p = 0.f;
  for (int c = threadIdx.x; c < DIM; c += 256) p += xr[c];
  const float mu = blockRedSum(p, red) * (1.f / DIM);
  float q = 0.f;
  for (int c = threadIdx.x; c < DIM; c += 256) { float d = xr[c] - mu; q += d * d; }
  const float inv = rsqrtf(blockRedSum(q, red) * (1.f / DIM) + LN_EPS);
  for (int c = threadIdx.x; c < DIM; c += 256)
    a_s[c] = (xr[c] - mu) * inv * ln1w[c] + ln1b[c];
  __syncthreads();
  for (int o = wave; o < DIM; o += 8) {
    const float* wr = qkv_w + (size_t)o * DIM;   // Q rows 0..383
    float s = 0.f;
#pragma unroll
    for (int k = 0; k < 12; ++k) { int c = lane + 32 * k; s += a_s[c] * wr[c]; }
    s = waveRedSum(s);
    if (lane == 0) qcls[b * DIM + o] = s;
  }
}

// ---------------- main kernel: LN1 + bf16 K/V WMMA GEMM + logits + residual out ----------------
__global__ __launch_bounds__(256) void kv_attn_kernel(
    const float* __restrict__ x, const float* __restrict__ ln1w,
    const float* __restrict__ ln1b, const float* __restrict__ g1,
    const __bf16* __restrict__ Wkv, const float* __restrict__ qcls,
    float* __restrict__ logits, __bf16* __restrict__ Vws,
    float* __restrict__ out) {
  __shared__ __bf16 aLDS[TILE * LDA];
  const int b = blockIdx.y;
  const int n0 = blockIdx.x * TILE;
  const int wave = threadIdx.x >> 5;
  const int lane = threadIdx.x & 31;

  // ---- Stage 1: LayerNorm 64 tokens (8 rows/wave), emit bf16 A-tile + residual out ----
  {
    const int c0 = lane * 12;
    float w12[12], b12[12], gg12[12];
#pragma unroll
    for (int j = 0; j < 12; ++j) { w12[j] = ln1w[c0 + j]; b12[j] = ln1b[c0 + j]; gg12[j] = g1[c0 + j]; }
    for (int rr = 0; rr < 8; ++rr) {
      const int r = wave * 8 + rr;
      const int n = n0 + r;
      if (n < NTOK) {
        float xv[12];
        const float* xr = x + ((size_t)b * NTOK + n) * DIM + c0;
#pragma unroll
        for (int k = 0; k < 3; ++k) {
          float4 t = *(const float4*)(xr + 4 * k);
          xv[4 * k] = t.x; xv[4 * k + 1] = t.y; xv[4 * k + 2] = t.z; xv[4 * k + 3] = t.w;
        }
        float s = 0.f;
#pragma unroll
        for (int j = 0; j < 12; ++j) s += xv[j];
        s = waveRedSum(s);
        const float mu = s * (1.f / DIM);
        float q = 0.f;
#pragma unroll
        for (int j = 0; j < 12; ++j) { float d = xv[j] - mu; q += d * d; }
        q = waveRedSum(q);
        const float inv = rsqrtf(q * (1.f / DIM) + LN_EPS);
        float a[12];
#pragma unroll
        for (int j = 0; j < 12; ++j) {
          a[j] = (xv[j] - mu) * inv * w12[j] + b12[j];
          aLDS[r * LDA + c0 + j] = (__bf16)a[j];
        }
        if (n >= 1) {  // non-class residual: out = 2*(x + gamma1*LN1(x)); token 0 done by tail
          float* orow = out + ((size_t)b * NTOK + n) * DIM + c0;
#pragma unroll
          for (int k = 0; k < 3; ++k) {
            float4 t;
            t.x = 2.f * (xv[4 * k]     + gg12[4 * k]     * a[4 * k]);
            t.y = 2.f * (xv[4 * k + 1] + gg12[4 * k + 1] * a[4 * k + 1]);
            t.z = 2.f * (xv[4 * k + 2] + gg12[4 * k + 2] * a[4 * k + 2]);
            t.w = 2.f * (xv[4 * k + 3] + gg12[4 * k + 3] * a[4 * k + 3]);
            *(float4*)(orow + 4 * k) = t;
          }
        }
      } else {
#pragma unroll
        for (int j = 0; j < 12; ++j) aLDS[r * LDA + c0 + j] = (__bf16)0.f;
      }
    }
  }
  __syncthreads();

  // ---- Stage 2: WMMA GEMM over 48 output tiles of Wkv (waves 0-3: K heads, 4-7: V) ----
  const int lhalf = lane >> 4;   // 0|1
  const int lm = lane & 15;
  // LDS byte address of this lane's A-fragment base (low 32 bits of generic ptr = LDS offset)
  const unsigned int aAddrBase =
      (unsigned int)(uintptr_t)(&aLDS[0]) + (unsigned int)(lm * (LDA * 2) + lhalf * 16);

  auto loadB = [&](int t, int kc) -> v16bf {
    // 16-bit B 32x16 layout: lane=N, lanes<16 K=0..7,16..23, lanes>=16 K=8..15,24..31
    const __bf16* p = Wkv + ((size_t)(t * 16 + lm)) * DIM + kc * 32 + 8 * lhalf;
    v8bf lo = *(const v8bf*)p;
    v8bf hi = *(const v8bf*)(p + 16);
    return __builtin_shufflevector(lo, hi, 0,1,2,3,4,5,6,7,8,9,10,11,12,13,14,15);
  };

  // one o-tile GEMM: full kc unroll, 1-deep B prefetch, A pinned to LDS via loadA4
#define OTILE_GEMM(T)                                                     \
  {                                                                       \
    v16bf bcur = loadB((T), 0);                                           \
    _Pragma("unroll")                                                     \
    for (int kc = 0; kc < 11; ++kc) {                                     \
      v16bf bnxt = loadB((T), kc + 1);                                    \
      v16bf a0, a1, a2, a3;                                               \
      loadA4(aAddrBase + (unsigned int)(kc * 64), a0, a1, a2, a3);        \
      acc0 = WMMA_BF16(a0, bcur, acc0);                                   \
      acc1 = WMMA_BF16(a1, bcur, acc1);                                   \
      acc2 = WMMA_BF16(a2, bcur, acc2);                                   \
      acc3 = WMMA_BF16(a3, bcur, acc3);                                   \
      bcur = bnxt;                                                        \
    }                                                                     \
    v16bf a0, a1, a2, a3;                                                 \
    loadA4(aAddrBase + (unsigned int)(11 * 64), a0, a1, a2, a3);          \
    acc0 = WMMA_BF16(a0, bcur, acc0);                                     \
    acc1 = WMMA_BF16(a1, bcur, acc1);                                     \
    acc2 = WMMA_BF16(a2, bcur, acc2);                                     \
    acc3 = WMMA_BF16(a3, bcur, acc3);                                     \
  }

  if (wave < 4) {
    // K region: o-tiles 0..23; this wave owns heads 2w and 2w+1 (3 tiles each)
    for (int hh = 0; hh < 2; ++hh) {
      const int head = wave * 2 + hh;
      float plog[4][8];
#pragma unroll
      for (int mt = 0; mt < 4; ++mt)
#pragma unroll
        for (int rI = 0; rI < 8; ++rI) plog[mt][rI] = 0.f;
      for (int sub = 0; sub < 3; ++sub) {
        const int t = head * 3 + sub;
        const float q = qcls[b * DIM + t * 16 + lm];
        v8f acc0 = {}, acc1 = {}, acc2 = {}, acc3 = {};
        OTILE_GEMM(t)
#pragma unroll
        for (int rI = 0; rI < 8; ++rI) {
          plog[0][rI] += q * acc0[rI];
          plog[1][rI] += q * acc1[rI];
          plog[2][rI] += q * acc2[rI];
          plog[3][rI] += q * acc3[rI];
        }
      }
      float* lgrow = logits + ((size_t)(b * HEADS + head)) * NPAD + n0;
#pragma unroll
      for (int mt = 0; mt < 4; ++mt)
#pragma unroll
        for (int rI = 0; rI < 8; ++rI) {
          const float v = halfRedSum(plog[mt][rI]);
          if (lm == 0) {
            const int m0 = mt * 16 + rI + 8 * lhalf;   // token row in tile
            lgrow[m0] = ((n0 + m0) < NTOK) ? v * ATTN_SCALE : -1e30f;
          }
        }
    }
  } else {
    // V region: o-tiles 24..47 -> bf16 V spill
    for (int i = 0; i < 6; ++i) {
      const int t = 24 + (wave - 4) * 6 + i;
      v8f acc0 = {}, acc1 = {}, acc2 = {}, acc3 = {};
      OTILE_GEMM(t)
      const int ov = (t - 24) * 16 + lm;       // 0..383 within V
      __bf16* vb = Vws + ((size_t)b * NTOK + n0) * DIM + ov;
#pragma unroll
      for (int rI = 0; rI < 8; ++rI) {
        const int m0 = rI + 8 * lhalf;
        if ((n0 + m0) < NTOK)      vb[(size_t)m0 * DIM]        = (__bf16)acc0[rI];
        if ((n0 + 16 + m0) < NTOK) vb[(size_t)(16 + m0) * DIM] = (__bf16)acc1[rI];
        if ((n0 + 32 + m0) < NTOK) vb[(size_t)(32 + m0) * DIM] = (__bf16)acc2[rI];
        if ((n0 + 48 + m0) < NTOK) vb[(size_t)(48 + m0) * DIM] = (__bf16)acc3[rI];
      }
    }
  }
#undef OTILE_GEMM
}

// ---------------- kernel: softmax over N + weighted V sum -> ctx ----------------
__global__ __launch_bounds__(256) void softmax_ctx_kernel(const float* __restrict__ logits,
                                                          const __bf16* __restrict__ Vws,
                                                          float* __restrict__ ctx) {
  const int bh = blockIdx.x;
  const int b = bh >> 3, h = bh & 7;
  __shared__ float red[8];
  __shared__ float ctxsh[HD];
  const float* lg = logits + (size_t)bh * NPAD;
  const int wave = threadIdx.x >> 5, lane = threadIdx.x & 31;

  float m = -1e30f;
  for (int n = threadIdx.x; n < NTOK; n += 256) m = fmaxf(m, lg[n]);
#pragma unroll
  for (int mask = 16; mask >= 1; mask >>= 1) m = fmaxf(m, __shfl_xor(m, mask, 32));
  if (lane == 0) red[wave] = m;
  __syncthreads();
  if (threadIdx.x == 0) {
    float mm = red[0];
#pragma unroll
    for (int w = 1; w < 8; ++w) mm = fmaxf(mm, red[w]);
    red[0] = mm;
  }
  __syncthreads();
  const float M = red[0];
  __syncthreads();
  if (threadIdx.x < HD) ctxsh[threadIdx.x] = 0.f;
  __syncthreads();

  float acc[HD];
#pragma unroll
  for (int d = 0; d < HD; ++d) acc[d] = 0.f;
  float s = 0.f;
  for (int n = threadIdx.x; n < NTOK; n += 256) {
    const float e = __expf(lg[n] - M);
    s += e;
    const v8bf* vp = (const v8bf*)(Vws + ((size_t)b * NTOK + n) * DIM + h * HD);
#pragma unroll
    for (int g = 0; g < 6; ++g) {
      v8bf vv = vp[g];
#pragma unroll
      for (int e2 = 0; e2 < 8; ++e2) acc[g * 8 + e2] += e * (float)vv[e2];
    }
  }
  s = waveRedSum(s);
  if (lane == 0) red[wave] = s;
  __syncthreads();
  if (threadIdx.x == 0) {
    float ss = 0.f;
#pragma unroll
    for (int w = 0; w < 8; ++w) ss += red[w];
    red[0] = ss;
  }
  __syncthreads();
  const float S = red[0];
#pragma unroll
  for (int d = 0; d < HD; ++d) {
    const float v = waveRedSum(acc[d]);
    if (lane == 0) atomicAdd(&ctxsh[d], v);
  }
  __syncthreads();
  if (threadIdx.x < HD) ctx[(size_t)bh * HD + threadIdx.x] = ctxsh[threadIdx.x] / S;
}

// ---------------- kernel: class-token tail (proj + LN2 + MLP), lane-parallel dots ----------------
__global__ __launch_bounds__(256) void cls_tail_kernel(
    const float* __restrict__ x, const float* __restrict__ ctx,
    const float* __restrict__ proj_w, const float* __restrict__ proj_b,
    const float* __restrict__ g1, const float* __restrict__ ln2w,
    const float* __restrict__ ln2b, const float* __restrict__ fc1w,
    const float* __restrict__ fc1b, const float* __restrict__ fc2w,
    const float* __restrict__ fc2b, const float* __restrict__ g2,
    float* __restrict__ out) {
  const int b = blockIdx.x;
  __shared__ float ctx_s[DIM];
  __shared__ float t_s[DIM];
  __shared__ float h_s[HIDDEN];
  __shared__ float red[8];
  const int wave = threadIdx.x >> 5;
  const int lane = threadIdx.x & 31;
  for (int c = threadIdx.x; c < DIM; c += 256) ctx_s[c] = ctx[(size_t)b * DIM + c];
  __syncthreads();
  // proj + gamma1 residual -> x1[:,0]
  for (int o = wave; o < DIM; o += 8) {
    const float* wr = proj_w + (size_t)o * DIM;
    float s = 0.f;
#pragma unroll
    for (int k = 0; k < 12; ++k) { int c = lane + 32 * k; s += ctx_s[c] * wr[c]; }
    s = waveRedSum(s) + proj_b[o];
    if (lane == 0) t_s[o] = x[((size_t)b * NTOK) * DIM + o] + g1[o] * s;
  }
  __syncthreads();
  // LN2
  float p = 0.f;
  for (int c = threadIdx.x; c < DIM; c += 256) p += t_s[c];
  const float mu = blockRedSum(p, red) * (1.f / DIM);
  float q = 0.f;
  for (int c = threadIdx.x; c < DIM; c += 256) { float d = t_s[c] - mu; q += d * d; }
  const float inv = rsqrtf(blockRedSum(q, red) * (1.f / DIM) + LN_EPS);
  for (int c = threadIdx.x; c < DIM; c += 256)
    t_s[c] = (t_s[c] - mu) * inv * ln2w[c] + ln2b[c];
  __syncthreads();
  // fc1 + exact GELU
  for (int j = wave; j < HIDDEN; j += 8) {
    const float* wr = fc1w + (size_t)j * DIM;
    float s = 0.f;
#pragma unroll
    for (int k = 0; k < 12; ++k) { int c = lane + 32 * k; s += t_s[c] * wr[c]; }
    s = waveRedSum(s) + fc1b[j];
    if (lane == 0) h_s[j] = 0.5f * s * (1.f + erff(s * 0.70710678118654752f));
  }
  __syncthreads();
  // fc2 + gamma2 + residual -> out[:,0]
  for (int o = wave; o < DIM; o += 8) {
    const float* wr = fc2w + (size_t)o * HIDDEN;
    float s = 0.f;
#pragma unroll
    for (int k = 0; k < 48; ++k) { int j = lane + 32 * k; s += h_s[j] * wr[j]; }
    s = waveRedSum(s) + fc2b[o];
    if (lane == 0)
      out[((size_t)b * NTOK) * DIM + o] = t_s[o] + g2[o] * s;
  }
}

// ---------------- launcher ----------------
extern "C" void kernel_launch(void* const* d_in, const int* in_sizes, int n_in,
                              void* d_out, int out_size, void* d_ws, size_t ws_size,
                              hipStream_t stream) {
  (void)in_sizes; (void)n_in; (void)out_size; (void)ws_size;
  const float* x      = (const float*)d_in[0];
  const float* ln1w   = (const float*)d_in[3];
  const float* ln1b   = (const float*)d_in[4];
  const float* qkv_w  = (const float*)d_in[5];
  const float* proj_w = (const float*)d_in[6];
  const float* proj_b = (const float*)d_in[7];
  const float* ln2w   = (const float*)d_in[8];
  const float* ln2b   = (const float*)d_in[9];
  const float* fc1w   = (const float*)d_in[10];
  const float* fc1b   = (const float*)d_in[11];
  const float* fc2w   = (const float*)d_in[12];
  const float* fc2b   = (const float*)d_in[13];
  const float* g1     = (const float*)d_in[14];
  const float* g2     = (const float*)d_in[15];
  float* out = (float*)d_out;

  char* ws = (char*)d_ws;
  const size_t offWkv    = 0;                                   // 768*384*2   = 589824 B
  const size_t offQcls   = offWkv  + (size_t)768 * DIM * 2;     // 16*384*4    = 24576 B
  const size_t offLogits = offQcls + (size_t)BATCH * DIM * 4;   // 16*8*4160*4 = 2129920 B
  const size_t offCtx    = offLogits + (size_t)BATCH * HEADS * NPAD * 4;
  const size_t offV      = offCtx  + (size_t)BATCH * DIM * 4;   // 16*4097*384*2 ~ 50.3 MB
  __bf16* Wkv   = (__bf16*)(ws + offWkv);
  float*  qcls  = (float*)(ws + offQcls);
  float*  lgts  = (float*)(ws + offLogits);
  float*  ctx   = (float*)(ws + offCtx);
  __bf16* Vws   = (__bf16*)(ws + offV);

  convw_kernel<<<(768 * DIM + 255) / 256, 256, 0, stream>>>(qkv_w, Wkv);
  qcls_kernel<<<BATCH, 256, 0, stream>>>(x, ln1w, ln1b, qkv_w, qcls);
  dim3 grid1(NTILES, BATCH);
  kv_attn_kernel<<<grid1, 256, 0, stream>>>(x, ln1w, ln1b, g1, Wkv, qcls, lgts, Vws, out);
  softmax_ctx_kernel<<<BATCH * HEADS, 256, 0, stream>>>(lgts, Vws, ctx);
  cls_tail_kernel<<<BATCH, 256, 0, stream>>>(x, ctx, proj_w, proj_b, g1, ln2w, ln2b,
                                             fc1w, fc1b, fc2w, fc2b, g2, out);
}
